// EnhancedMultHeadAttention_34291018891499
// MI455X (gfx1250) — matrix operand order repaired and checked by reading, not compiled, not run on hardware
//
#include <hip/hip_runtime.h>
#include <hip/hip_bf16.h>

// ---------------------------------------------------------------------------
// CDNA5 / gfx1250 WMMA attention pipeline (wave32, v_wmma_f32_16x16x32_bf16,
// async global->LDS staging when the toolchain exposes it)
// ---------------------------------------------------------------------------

typedef __attribute__((ext_vector_type(16))) __bf16 v16bf;
typedef __attribute__((ext_vector_type(8)))  float  v8f;
typedef __attribute__((ext_vector_type(4)))  int    v4i;

static constexpr int BM   = 128;  // block tile M
static constexpr int BN   = 128;  // block tile N
static constexpr int BK   = 32;   // K per WMMA / per LDS stage
static constexpr int TPB  = 256;  // 8 waves of 32
static constexpr int LDSS = 40;   // padded LDS row stride (32+8) -> 80B, conflict-free b128

union Frag { v16bf v; uint4 q[2]; };

#if __has_builtin(__builtin_amdgcn_global_load_async_to_lds_b128)
#define ASYNC_CP 1
typedef __attribute__((address_space(1))) v4i  gbl_v4i;
typedef __attribute__((address_space(3))) v4i  lds_v4i;
typedef __attribute__((address_space(1))) void gbl_void;
typedef __attribute__((address_space(3))) void lds_void;
__device__ __forceinline__ void async_cp16(const void* g, void* l) {
  // global -> LDS, 16B per lane, tracked by ASYNCcnt (no VGPR landing zone)
  __builtin_amdgcn_global_load_async_to_lds_b128(
      (gbl_v4i*)(gbl_void*)g,
      (lds_v4i*)(lds_void*)l, 0, 0);
}
__device__ __forceinline__ void wait_async0() {
#if __has_builtin(__builtin_amdgcn_s_wait_asynccnt)
  __builtin_amdgcn_s_wait_asynccnt(0);
#else
  asm volatile("s_wait_asynccnt 0x0" ::: "memory");
#endif
}
#endif

// C[m,n] = scale * sum_k A[m,k]*B[n,k] + bias[n]        (NT GEMM, batched)
// A: bf16 (or f32 converted on load), B: bf16 row-major [N x K].
// FULL = true when M % BM == 0 and N % BN == 0 (no bounds checks anywhere).
template<bool AF32, bool OUTF32, bool FULL>
__global__ __launch_bounds__(TPB)
void gemm_nt_wmma(const void* __restrict__ Ap,
                  const __hip_bfloat16* __restrict__ Bp,
                  const float* __restrict__ bias,
                  void* __restrict__ Cp,
                  int M, int N, int K, int lda, int ldb, int ldc, float scale,
                  int innerDiv,
                  long long aSO, long long aSI,
                  long long bSO, long long bSI,
                  long long cSO, long long cSI)
{
  __shared__ __hip_bfloat16 sA[2][BM * LDSS];
  __shared__ __hip_bfloat16 sB[2][BN * LDSS];

  const int tid = threadIdx.x;
  const int z   = blockIdx.z;
  const int zo  = z / innerDiv;
  const int zi  = z - zo * innerDiv;

  const float*          Af = (const float*)Ap          + (AF32 ? (zo * aSO + zi * aSI) : 0);
  const __hip_bfloat16* Ah = (const __hip_bfloat16*)Ap + (AF32 ? 0 : (zo * aSO + zi * aSI));
  const __hip_bfloat16* Bb = Bp + zo * bSO + zi * bSI;

  const int tileM = blockIdx.y * BM;
  const int tileN = blockIdx.x * BN;

  // Per-thread staging coordinates: 2 slots of 8 elements (16B) each.
  int rS[2], cS[2];
#pragma unroll
  for (int it = 0; it < 2; ++it) {
    const int slot = tid + it * TPB;
    rS[it] = slot >> 2;
    cS[it] = (slot & 3) * 8;
  }

  uint4 pa[2], pb[2];  // pending register-staged data (f32 path / non-async fallback)

  // ---- staging helpers: issue (before compute) / commit (after compute) ----
  auto issueA = [&](int buf, int k0) {
#pragma unroll
    for (int it = 0; it < 2; ++it) {
      const int  grow = tileM + rS[it];
      const bool ok   = FULL || (grow < M);
      if (AF32) {
        uint4 val;
        if (ok) {
          const float* gp = Af + (size_t)grow * lda + k0 + cS[it];
          const float4 f0 = *(const float4*)gp;
          const float4 f1 = *(const float4*)(gp + 4);
          union { uint4 q; __hip_bfloat16 h[8]; } u;
          u.h[0] = __float2bfloat16(f0.x); u.h[1] = __float2bfloat16(f0.y);
          u.h[2] = __float2bfloat16(f0.z); u.h[3] = __float2bfloat16(f0.w);
          u.h[4] = __float2bfloat16(f1.x); u.h[5] = __float2bfloat16(f1.y);
          u.h[6] = __float2bfloat16(f1.z); u.h[7] = __float2bfloat16(f1.w);
          val = u.q;
          if (k0 + 2 * BK < K) __builtin_prefetch((const void*)(gp + 2 * BK), 0, 1);
        } else { val.x = val.y = val.z = val.w = 0u; }
        pa[it] = val;
      } else {
        const __hip_bfloat16* gp = Ah + (size_t)grow * lda + k0 + cS[it];
        __hip_bfloat16*       sp = &sA[buf][rS[it] * LDSS + cS[it]];
#ifdef ASYNC_CP
        if (ok) {
          async_cp16(gp, sp);
          if (k0 + 2 * BK < K) __builtin_prefetch((const void*)(gp + 2 * BK), 0, 1);
        } else { uint4 zz; zz.x = zz.y = zz.z = zz.w = 0u; *(uint4*)sp = zz; }
#else
        uint4 val;
        if (ok) {
          val = *(const uint4*)gp;
          if (k0 + 2 * BK < K) __builtin_prefetch((const void*)(gp + 2 * BK), 0, 1);
        } else { val.x = val.y = val.z = val.w = 0u; }
        pa[it] = val;
        (void)sp;
#endif
      }
    }
  };
  auto commitA = [&](int buf) {
#ifdef ASYNC_CP
    if (!AF32) return;  // async loads already targeted LDS
#endif
#pragma unroll
    for (int it = 0; it < 2; ++it)
      *(uint4*)&sA[buf][rS[it] * LDSS + cS[it]] = pa[it];
  };
  auto issueB = [&](int buf, int k0) {
#pragma unroll
    for (int it = 0; it < 2; ++it) {
      const int  grow = tileN + rS[it];
      const bool ok   = FULL || (grow < N);
      const __hip_bfloat16* gp = Bb + (size_t)grow * ldb + k0 + cS[it];
      __hip_bfloat16*       sp = &sB[buf][rS[it] * LDSS + cS[it]];
#ifdef ASYNC_CP
      if (ok) {
        async_cp16(gp, sp);
        if (k0 + 2 * BK < K) __builtin_prefetch((const void*)(gp + 2 * BK), 0, 1);
      } else { uint4 zz; zz.x = zz.y = zz.z = zz.w = 0u; *(uint4*)sp = zz; }
#else
      uint4 val;
      if (ok) {
        val = *(const uint4*)gp;
        if (k0 + 2 * BK < K) __builtin_prefetch((const void*)(gp + 2 * BK), 0, 1);
      } else { val.x = val.y = val.z = val.w = 0u; }
      pb[it] = val;
      (void)sp;
#endif
    }
  };
  auto commitB = [&](int buf) {
#ifdef ASYNC_CP
    (void)buf;
#else
#pragma unroll
    for (int it = 0; it < 2; ++it)
      *(uint4*)&sB[buf][rS[it] * LDSS + cS[it]] = pb[it];
#endif
  };
  auto stage_fence = [&]() {
#ifdef ASYNC_CP
    wait_async0();  // drain ASYNCcnt before the workgroup barrier
#endif
  };

  // ---- compute setup ----
  const int lane = tid & 31;
  const int wave = tid >> 5;
  const int wm   = (wave & 1) * 64;   // 2 waves along M (64 rows each)
  const int wn   = (wave >> 1) * 32;  // 4 waves along N (32 cols each)
  const int lr   = lane & 15;
  const int hh   = lane >> 4;         // half-wave select per ISA 7.12.2 layouts

  v8f acc[4][2];
#pragma unroll
  for (int t = 0; t < 4; ++t)
#pragma unroll
    for (int u = 0; u < 2; ++u)
#pragma unroll
      for (int r = 0; r < 8; ++r) acc[t][u][r] = 0.0f;

  const int nstages = K / BK;

  issueA(0, 0); issueB(0, 0);
  commitA(0);   commitB(0);
  stage_fence();
  __syncthreads();

  for (int s = 0; s < nstages; ++s) {
    const int  cur  = s & 1;
    const bool more = (s + 1 < nstages);
    if (more) {                       // issue next tile, then hide it behind WMMAs
      issueA(cur ^ 1, (s + 1) * BK);
      issueB(cur ^ 1, (s + 1) * BK);
    }

    // A fragment (16x32): lane holds row M=lr; low half-wave K=0..7,16..23,
    // high half-wave K=8..15,24..31.  B fragment (32x16): lane holds col N=lr;
    // low half K=0..7,16..23, high half K=8..15,24..31.
    Frag a[4], b[2];
#pragma unroll
    for (int t = 0; t < 4; ++t) {
      const __hip_bfloat16* p = &sA[cur][(wm + t * 16 + lr) * LDSS + hh * 8];
      a[t].q[0] = *(const uint4*)(p);
      a[t].q[1] = *(const uint4*)(p + 16);
    }
#pragma unroll
    for (int u = 0; u < 2; ++u) {
      const __hip_bfloat16* p = &sB[cur][(wn + u * 16 + lr) * LDSS + hh * 8];
      b[u].q[0] = *(const uint4*)(p);
      b[u].q[1] = *(const uint4*)(p + 16);
    }

#pragma unroll
    for (int t = 0; t < 4; ++t)
#pragma unroll
      for (int u = 0; u < 2; ++u)
        acc[t][u] = __builtin_amdgcn_wmma_f32_16x16x32_bf16(
            false, a[t].v, false, b[u].v, (short)0, acc[t][u], false, false);

    if (more) { commitA(cur ^ 1); commitB(cur ^ 1); stage_fence(); }
    __syncthreads();
  }

  // ---- epilogue: v8f VGPR r <-> M = base + r + 8*half; N = base + (lane&15) ----
  const long long coff = zo * cSO + zi * cSI;
#pragma unroll
  for (int t = 0; t < 4; ++t) {
    const int rbase = tileM + wm + t * 16 + hh * 8;
#pragma unroll
    for (int u = 0; u < 2; ++u) {
      const int col = tileN + wn + u * 16 + lr;
      if (FULL || col < N) {
        const float bv = bias ? bias[col] : 0.0f;
#pragma unroll
        for (int r = 0; r < 8; ++r) {
          const int row = rbase + r;
          if (FULL || row < M) {
            const float val = acc[t][u][r] * scale + bv;
            if (OUTF32)
              ((float*)Cp)[coff + (size_t)row * ldc + col] = val;
            else
              ((__hip_bfloat16*)Cp)[coff + (size_t)row * ldc + col] =
                  __float2bfloat16(val);
          }
        }
      }
    }
  }
}

// ---------------------------------------------------------------------------

__global__ void cvt_f32_to_bf16(const float* __restrict__ in,
                                __hip_bfloat16* __restrict__ out, int n)
{
  const int i = (blockIdx.x * TPB + threadIdx.x) * 4;
  if (i + 3 < n) {
    const float4 f = *(const float4*)(in + i);
    out[i + 0] = __float2bfloat16(f.x);
    out[i + 1] = __float2bfloat16(f.y);
    out[i + 2] = __float2bfloat16(f.z);
    out[i + 3] = __float2bfloat16(f.w);
  } else {
    for (int j = i; j < n; ++j) out[j] = __float2bfloat16(in[j]);
  }
}

// v_p [B*S, D] -> vpt [B*H, DEPTH, S]  so logits @ v becomes an NT GEMM.
__global__ void head_transpose_bf16(const __hip_bfloat16* __restrict__ in,
                                    __hip_bfloat16* __restrict__ out)
{
  const int i    = blockIdx.x * TPB + threadIdx.x;  // over B*S*D
  const int dcol = i & 1023;
  const int bs   = i >> 10;
  const int t    = bs & 1023;
  const int bb   = bs >> 10;
  const int h    = dcol >> 6;
  const int d    = dcol & 63;
  out[((size_t)((bb * 16 + h) * 64 + d) << 10) + t] = in[i];
}

// In-place softmax over rows of 1024 f32 (one block per row).
__global__ __launch_bounds__(TPB) void softmax_rows_1024(float* __restrict__ data)
{
  float* row = data + ((size_t)blockIdx.x << 10);
  __shared__ float red[TPB];
  const int tid = threadIdx.x;
  float v0 = row[tid], v1 = row[tid + 256], v2 = row[tid + 512], v3 = row[tid + 768];
  float m = fmaxf(fmaxf(v0, v1), fmaxf(v2, v3));
  red[tid] = m; __syncthreads();
  for (int s = 128; s > 0; s >>= 1) {
    if (tid < s) red[tid] = fmaxf(red[tid], red[tid + s]);
    __syncthreads();
  }
  m = red[0]; __syncthreads();
  v0 = __expf(v0 - m); v1 = __expf(v1 - m); v2 = __expf(v2 - m); v3 = __expf(v3 - m);
  red[tid] = v0 + v1 + v2 + v3; __syncthreads();
  for (int s = 128; s > 0; s >>= 1) {
    if (tid < s) red[tid] += red[tid + s];
    __syncthreads();
  }
  const float inv = 1.0f / red[0];
  row[tid] = v0 * inv; row[tid + 256] = v1 * inv;
  row[tid + 512] = v2 * inv; row[tid + 768] = v3 * inv;
}

// ---------------------------------------------------------------------------

extern "C" void kernel_launch(void* const* d_in, const int* in_sizes, int n_in,
                              void* d_out, int out_size, void* d_ws, size_t ws_size,
                              hipStream_t stream)
{
  (void)in_sizes; (void)n_in; (void)out_size; (void)ws_size;
  constexpr int B = 4, S = 1024, D = 1024, H = 16, DH = 64;
  constexpr size_t SD = (size_t)B * S * D;   // 4,194,304
  constexpr size_t WW = (size_t)D * D;       // 1,048,576

  const float* v_f  = (const float*)d_in[0];
  // d_in[1] (k) unused — reference overwrites k with wv(v)
  const float* q_f  = (const float*)d_in[2];
  const float* wq_w = (const float*)d_in[3];
  const float* wq_b = (const float*)d_in[4];
  const float* wv_w = (const float*)d_in[5];
  const float* wv_b = (const float*)d_in[6];
  const float* dn_w = (const float*)d_in[7];
  const float* dn_b = (const float*)d_in[8];

  __hip_bfloat16* ws    = (__hip_bfloat16*)d_ws;   // ~62 MiB total
  __hip_bfloat16* q_bf  = ws;
  __hip_bfloat16* v_bf  = ws + SD;
  __hip_bfloat16* qp_bf = ws + 2 * SD;
  __hip_bfloat16* vp_bf = ws + 3 * SD;
  __hip_bfloat16* kp_bf = ws + 4 * SD;
  __hip_bfloat16* vpt   = ws + 5 * SD;             // [B*H, DH, S]
  __hip_bfloat16* at_bf = ws + 6 * SD;             // concat(attn_out) [B*S, D]
  __hip_bfloat16* wq_bf = ws + 7 * SD;
  __hip_bfloat16* wv_bf = ws + 7 * SD + WW;
  __hip_bfloat16* dn_bf = ws + 7 * SD + 2 * WW;

  float* outO = (float*)d_out;        // output       [B,S,D]
  float* outA = (float*)d_out + SD;   // attn weights [B,H,S,S]

  const dim3 blk(TPB);

  cvt_f32_to_bf16<<<(unsigned)(SD / 4 / TPB), blk, 0, stream>>>(q_f, q_bf, (int)SD);
  cvt_f32_to_bf16<<<(unsigned)(SD / 4 / TPB), blk, 0, stream>>>(v_f, v_bf, (int)SD);
  cvt_f32_to_bf16<<<(unsigned)(WW / 4 / TPB), blk, 0, stream>>>(wq_w, wq_bf, (int)WW);
  cvt_f32_to_bf16<<<(unsigned)(WW / 4 / TPB), blk, 0, stream>>>(wv_w, wv_bf, (int)WW);
  cvt_f32_to_bf16<<<(unsigned)(WW / 4 / TPB), blk, 0, stream>>>(dn_w, dn_bf, (int)WW);

  // q_p = q @ wq^T + wq_b ; v_p = v @ wv^T + wv_b ; k_p = v_p @ wv^T + wv_b
  gemm_nt_wmma<false, false, true><<<dim3(D / BN, B * S / BM, 1), blk, 0, stream>>>(
      q_bf, wq_bf, wq_b, qp_bf, B * S, D, D, D, D, D, 1.0f, 1, 0, 0, 0, 0, 0, 0);
  gemm_nt_wmma<false, false, true><<<dim3(D / BN, B * S / BM, 1), blk, 0, stream>>>(
      v_bf, wv_bf, wv_b, vp_bf, B * S, D, D, D, D, D, 1.0f, 1, 0, 0, 0, 0, 0, 0);
  gemm_nt_wmma<false, false, true><<<dim3(D / BN, B * S / BM, 1), blk, 0, stream>>>(
      vp_bf, wv_bf, wv_b, kp_bf, B * S, D, D, D, D, D, 1.0f, 1, 0, 0, 0, 0, 0, 0);

  // logits[b,h] = (qh @ kh^T)/8  -> raw f32 straight into d_out attention region
  gemm_nt_wmma<false, true, true><<<dim3(S / BN, S / BM, B * H), blk, 0, stream>>>(
      qp_bf, kp_bf, nullptr, outA, S, S, DH, D, D, S, 0.125f,
      H, (long long)S * D, DH, (long long)S * D, DH,
      (long long)H * S * S, (long long)S * S);

  // vh^T per head, so attn_out is also an NT GEMM
  head_transpose_bf16<<<(unsigned)(SD / TPB), blk, 0, stream>>>(vp_bf, vpt);

  // attn_out[b,h] = logits @ vh   (A = raw f32 logits, converted on load; N=64 -> guards)
  gemm_nt_wmma<true, false, false><<<dim3(1, S / BM, B * H), blk, 0, stream>>>(
      outA, vpt, nullptr, at_bf, S, DH, S, S, S, D, 1.0f,
      H, (long long)H * S * S, (long long)S * S,
      (long long)H * DH * S, (long long)DH * S,
      (long long)S * D, DH);

  // output = concat @ dense^T + dense_b
  gemm_nt_wmma<false, true, true><<<dim3(D / BN, B * S / BM, 1), blk, 0, stream>>>(
      at_bf, dn_bf, dn_b, outO, B * S, D, D, D, D, D, 1.0f, 1, 0, 0, 0, 0, 0, 0);

  // attention_weights = softmax(logits), in place (after attn GEMM consumed raw logits)
  softmax_rows_1024<<<B * H * S, blk, 0, stream>>>(outA);
}
